// ConvMGSA_74612171866139
// MI455X (gfx1250) — matrix operand order repaired and checked by reading, not compiled, not run on hardware
//
#include <hip/hip_runtime.h>
#include <hip/hip_bf16.h>
#include <stdint.h>

// ConvMGSA for MI455X (gfx1250): fp32 in/out, f16 WMMA GEMMs (fp32 accumulate),
// async-to-LDS weight staging (ASYNCcnt path), VALU-fused attention (K/V fits
// in 192MB L2), wave-cooperative LN writing the reshape-scramble layout.

typedef __attribute__((ext_vector_type(16))) _Float16 v16h;
typedef __attribute__((ext_vector_type(8)))  float    v8f;

#define CIN  256
#define COUT 256
#define GG   8
#define NN   4
#define HH   32
#define WWp  32
#define PP   1024   // H*W
#define MMh  4      // heads
#define DD   64     // COUT/M

union FragH { v16h h; uint32_t u[8]; };

// ---------------------------------------------------------------------------
// Grouped 1x1-conv GEMM:  Y[nb, co, p] = sum_ci W[g, co, ci] * X[nb, ci, p]
// Block tile 128(M) x 128(N), 8 waves, wave tile 64x32.
// A (weights) staged fp32 via GLOBAL_LOAD_ASYNC_TO_LDS_B128 (no VGPR round
// trip), converted to f16 at fragment build.  B staged via VGPR path because
// it is transposed (pixel-major -> [n][k]) and converted to f16 during store.
// ---------------------------------------------------------------------------
__global__ __launch_bounds__(256)
void gconv_gemm(const float* __restrict__ X, const float* __restrict__ Wt,
                const float* __restrict__ bias, float* __restrict__ Y, int relu)
{
    __shared__ float    Asf[128 * 36];  // [m][k] fp32, pitch 36 dwords (16B-aligned rows)
    __shared__ _Float16 Bs [128 * 34];  // [n][k] f16, pitch 34 halves (conflict-free)

    const int nb = blockIdx.z;
    const int g  = nb & 7;
    const int m0 = blockIdx.y * 128;
    const int n0 = blockIdx.x * 128;
    const size_t xbase = (size_t)nb * CIN  * PP;
    const size_t wbase = (size_t)g  * COUT * CIN;
    const size_t ybase = (size_t)nb * COUT * PP;

    const int tid   = threadIdx.x;
    const int lane  = tid & 31;
    const int wave  = tid >> 5;
    const int wmOff = (wave >> 2) * 64;  // 0 / 64
    const int wnOff = (wave & 3) * 32;   // 0..96
    const int r     = lane & 15;
    const int hi    = lane >> 4;

    const uint32_t as_base = (uint32_t)(uintptr_t)(&Asf[0]);  // LDS byte offset

    v8f zero = {};
    v8f acc[4][2];
#pragma unroll
    for (int mt = 0; mt < 4; ++mt)
#pragma unroll
        for (int nt = 0; nt < 2; ++nt) acc[mt][nt] = zero;

    for (int kt = 0; kt < CIN; kt += 32) {
        __syncthreads();   // prev-iter LDS reads complete (s_wait_dscnt + barrier)

        // --- A tile: 128x32 fp32 = 1024 x 16B chunks, DMA'd straight to LDS.
#pragma unroll
        for (int e = 0; e < 4; ++e) {
            int c   = tid + e * 256;
            int row = c >> 3, cx = c & 7;
            const float* gp = &Wt[wbase + (size_t)(m0 + row) * CIN + kt + cx * 4];
            uint32_t lds = as_base + (uint32_t)((row * 36 + cx * 4) * 4);
            asm volatile("global_load_async_to_lds_b128 %0, %1, off"
                         :: "v"(lds), "v"((uint64_t)(uintptr_t)gp)
                         : "memory");
        }

        // --- B tile: transpose+convert through VGPRs (coalesced global reads).
#pragma unroll
        for (int e = 0; e < 16; ++e) {
            int idx = tid + e * 256;
            int kk = idx >> 7, nn = idx & 127;
            Bs[nn * 34 + kk] = (_Float16)X[xbase + (size_t)(kt + kk) * PP + n0 + nn];
        }
        if (kt + 32 < CIN)
            __builtin_prefetch(&X[xbase + (size_t)(kt + 32 + (tid >> 3)) * PP + n0 + ((tid & 7) << 4)], 0, 3);

        asm volatile("s_wait_asynccnt 0x0" ::: "memory");  // A DMA landed in LDS
        __syncthreads();

        // --- Fragments per ISA 7.12.2 wave32 layouts.
        FragH fa[4], fb[2];
#pragma unroll
        for (int mt = 0; mt < 4; ++mt) {
            const float* ap = &Asf[(wmOff + mt * 16 + r) * 36];
#pragma unroll
            for (int p = 0; p < 8; ++p) {
                // A 16x32 f16: VGPR p holds K-pair {k,k+1}, k = 2p + (p>=4?8:0) + 8*hi
                int k = 2 * p + ((p >= 4) ? 8 : 0) + 8 * hi;
                float2 f = *(const float2*)(ap + k);          // ds_load_b64
                union { _Float16 h[2]; uint32_t u; } cv;
                cv.h[0] = (_Float16)f.x; cv.h[1] = (_Float16)f.y;
                fa[mt].u[p] = cv.u;
            }
        }
#pragma unroll
        for (int nt = 0; nt < 2; ++nt) {
            // B 32x16 f16: VGPR p holds K-pair {k,k+1}, k = 2p + (p>=4?8:0) + 8*hi (same as A)
            const _Float16* bp = &Bs[(wnOff + nt * 16 + r) * 34 + 8 * hi];
#pragma unroll
            for (int p = 0; p < 8; ++p)
                fb[nt].u[p] = *(const uint32_t*)(bp + 2 * p + ((p >= 4) ? 8 : 0));
        }
#pragma unroll
        for (int mt = 0; mt < 4; ++mt)
#pragma unroll
            for (int nt = 0; nt < 2; ++nt)
                acc[mt][nt] = __builtin_amdgcn_wmma_f32_16x16x32_f16(
                    false, fa[mt].h, false, fb[nt].h, (short)0, acc[mt][nt], false, false);
    }

    // Epilogue: C/D layout -> row = p + 8*hi, col = lane&15
#pragma unroll
    for (int mt = 0; mt < 4; ++mt) {
#pragma unroll
        for (int nt = 0; nt < 2; ++nt) {
            int colb = n0 + wnOff + nt * 16 + r;
            int rowb = m0 + wmOff + mt * 16 + 8 * hi;
#pragma unroll
            for (int p = 0; p < 8; ++p) {
                int row = rowb + p;
                float v = acc[mt][nt][p];
                if (bias) v += bias[g * COUT + row];
                if (relu) v = fmaxf(v, 0.f);
                Y[ybase + (size_t)row * PP + colb] = v;
            }
        }
    }
}

// ---------------------------------------------------------------------------
// Fused attention: scores (q . (k_shift + emb)) -> softmax(72) -> attn @ v.
// One thread per (n, m, g, pixel); 256 consecutive pixels per block so all
// channel-strided loads are coalesced across the wave. K0/V0 (67MB) live in L2.
// ---------------------------------------------------------------------------
__global__ __launch_bounds__(256)
void attn_fused(const float* __restrict__ Q, const float* __restrict__ K0,
                const float* __restrict__ V0, const float* __restrict__ hm,
                const float* __restrict__ wmr, float* __restrict__ AV)
{
    const int b  = blockIdx.x;            // 512 blocks = 4n * 4m * 8g * 4pt
    const int pt = b & 3;
    const int g  = (b >> 2) & 7;
    const int m  = (b >> 5) & 3;
    const int n  = b >> 7;
    const int p  = (pt << 8) + threadIdx.x;
    const int h  = p >> 5, w = p & 31;

    const size_t qoff = ((size_t)(n * 8 + g) * COUT + m * DD) * PP + p;
    float q[DD];
#pragma unroll
    for (int d = 0; d < DD; ++d) q[d] = Q[qoff + (size_t)d * PP];

    // Embedding dot-products: hm varies over i (e<4), wm over j (e>=4).
    float hdot[4][3], wdot[4][3];
    for (int e = 0; e < 4; ++e)
        for (int i = 0; i < 3; ++i) {
            const float* hp = hm + (size_t)(e * COUT + m * DD) * 3 + i;
            float s = 0.f;
#pragma unroll
            for (int d = 0; d < DD; ++d) s += q[d] * hp[d * 3];
            hdot[e][i] = s;
        }
    for (int e = 0; e < 4; ++e)
        for (int j = 0; j < 3; ++j) {
            const float* wp = wmr + (size_t)(e * COUT + m * DD) * 3 + j;
            float s = 0.f;
#pragma unroll
            for (int d = 0; d < DD; ++d) s += q[d] * wp[d * 3];
            wdot[e][j] = s;
        }

    float sc[72];
    for (int e = 0; e < 8; ++e) {
        const size_t kb = ((size_t)(n * 8 + e) * COUT + m * DD) * PP;
        for (int i = 0; i < 3; ++i) {
            int hh2 = h + i - 1;
            for (int j = 0; j < 3; ++j) {
                int ww2 = w + j - 1;
                float s = (e < 4) ? hdot[e][i] : wdot[e - 4][j];  // emb term survives padding
                if (hh2 >= 0 && hh2 < HH && ww2 >= 0 && ww2 < WWp) {
                    const float* kp = K0 + kb + hh2 * WWp + ww2;
#pragma unroll
                    for (int d = 0; d < DD; ++d) s += q[d] * kp[(size_t)d * PP];
                }
                sc[e * 9 + i * 3 + j] = s;
            }
        }
    }
    float mx = sc[0];
    for (int t = 1; t < 72; ++t) mx = fmaxf(mx, sc[t]);
    float sum = 0.f;
    for (int t = 0; t < 72; ++t) { float v = __expf(sc[t] - mx); sc[t] = v; sum += v; }
    const float inv = 1.f / sum;

    float out[DD];
#pragma unroll
    for (int d = 0; d < DD; ++d) out[d] = 0.f;
    for (int e = 0; e < 8; ++e) {
        const size_t vb = ((size_t)(n * 8 + e) * COUT + m * DD) * PP;
        for (int i = 0; i < 3; ++i) {
            int hh2 = h + i - 1; if (hh2 < 0 || hh2 >= HH) continue;
            for (int j = 0; j < 3; ++j) {
                int ww2 = w + j - 1; if (ww2 < 0 || ww2 >= WWp) continue;
                float a = sc[e * 9 + i * 3 + j] * inv;
                const float* vp = V0 + vb + hh2 * WWp + ww2;
#pragma unroll
                for (int d = 0; d < DD; ++d) out[d] += a * vp[(size_t)d * PP];
            }
        }
    }
#pragma unroll
    for (int d = 0; d < DD; ++d) AV[qoff + (size_t)d * PP] = out[d];
}

// ---------------------------------------------------------------------------
// t = LayerNorm_d(A + B) over d=64, written in (n, hw, m, g, d) order, which is
// exactly the reference's reshape-scrambled layout viewed as (n, g, 256, h, w).
// One wave32 per LN group (2 channels per lane), shfl_xor reductions.
// ---------------------------------------------------------------------------
__global__ __launch_bounds__(256)
void add_ln_scramble(const float* __restrict__ A, const float* __restrict__ B,
                     const float* __restrict__ gamma, const float* __restrict__ beta,
                     float* __restrict__ Out)
{
    const int wid  = threadIdx.x >> 5;
    const int lane = threadIdx.x & 31;
    const int gi   = blockIdx.x * 8 + wid;       // 2^17 groups: p[0:10) m[10:12) g[12:15) n[15:17)
    const int p  = gi & 1023;
    const int mq = (gi >> 10) & 3;
    const int g  = (gi >> 12) & 7;
    const int n  = gi >> 15;

    const size_t base = ((size_t)(n * 8 + g) * CIN + mq * DD) * PP + p;
    float v0 = A[base + (size_t)lane * PP]        + B[base + (size_t)lane * PP];
    float v1 = A[base + (size_t)(lane + 32) * PP] + B[base + (size_t)(lane + 32) * PP];

    float s = v0 + v1;
#pragma unroll
    for (int o = 16; o; o >>= 1) s += __shfl_xor(s, o, 32);
    const float mean = s * (1.f / 64.f);
    const float e0 = v0 - mean, e1 = v1 - mean;
    float vv = e0 * e0 + e1 * e1;
#pragma unroll
    for (int o = 16; o; o >>= 1) vv += __shfl_xor(vv, o, 32);
    const float inv = rsqrtf(vv * (1.f / 64.f) + 1e-5f);

    const size_t ob = ((size_t)n * PP + p) * 2048 + (size_t)mq * 512 + (size_t)g * 64;
    Out[ob + lane]      = e0 * inv * gamma[lane]      + beta[lane];
    Out[ob + lane + 32] = e1 * inv * gamma[lane + 32] + beta[lane + 32];
}

// ---------------------------------------------------------------------------
extern "C" void kernel_launch(void* const* d_in, const int* in_sizes, int n_in,
                              void* d_out, int out_size, void* d_ws, size_t ws_size,
                              hipStream_t stream)
{
    const float* x      = (const float*)d_in[0];
    const float* wq     = (const float*)d_in[1];
    const float* wk     = (const float*)d_in[2];
    const float* wv     = (const float*)d_in[3];
    const float* hm     = (const float*)d_in[4];
    const float* wmr    = (const float*)d_in[5];
    const float* w_conv = (const float*)d_in[6];
    const float* b_conv = (const float*)d_in[7];
    const float* w_f1   = (const float*)d_in[8];
    const float* b_f1   = (const float*)d_in[9];
    const float* w_f2   = (const float*)d_in[10];
    const float* b_f2   = (const float*)d_in[11];
    const float* g1     = (const float*)d_in[12];
    const float* b1     = (const float*)d_in[13];
    const float* g2     = (const float*)d_in[14];
    const float* b2     = (const float*)d_in[15];

    // 4 live fp32 tensors max (buffer recycling): 4 * 33.5MB = 128MB of d_ws.
    const size_t TENS = (size_t)NN * GG * CIN * PP;   // 8,388,608 floats
    float* Q  = (float*)d_ws;        // Q, later Y2
    float* Kb = Q  + TENS;           // K0, later T (scrambled LN output)
    float* Vb = Kb + TENS;           // V0, later F1
    float* Wb = Vb + TENS;           // AV, later F
    float* outp = (float*)d_out;

    dim3 ggrid(PP / 128, COUT / 128, NN * GG);
    const int ln_blocks = (NN * PP * MMh * GG) / 8;   // 16384

    gconv_gemm<<<ggrid, 256, 0, stream>>>(x, wq, nullptr, Q,  0);
    gconv_gemm<<<ggrid, 256, 0, stream>>>(x, wk, nullptr, Kb, 0);
    gconv_gemm<<<ggrid, 256, 0, stream>>>(x, wv, nullptr, Vb, 0);
    attn_fused<<<NN * MMh * GG * (PP / 256), 256, 0, stream>>>(Q, Kb, Vb, hm, wmr, Wb);
    gconv_gemm<<<ggrid, 256, 0, stream>>>(Wb, w_conv, b_conv, Q, 0);          // Y2 -> Q slot
    add_ln_scramble<<<ln_blocks, 256, 0, stream>>>(x, Q, g1, b1, Kb);         // T  -> Kb slot
    gconv_gemm<<<ggrid, 256, 0, stream>>>(Kb, w_f1, b_f1, Vb, 1);             // F1 -> Vb (ReLU)
    gconv_gemm<<<ggrid, 256, 0, stream>>>(Vb, w_f2, b_f2, Wb, 0);             // F  -> Wb
    add_ln_scramble<<<ln_blocks, 256, 0, stream>>>(Kb, Wb, g2, b2, outp);     // t2 -> d_out
}
